// MinimalRNNCell_3152505995317
// MI455X (gfx1250) — matrix operand order repaired and checked by reading, not compiled, not run on hardware
//
#include <hip/hip_runtime.h>
#include <hip/hip_bf16.h>

// ---------------------------------------------------------------------------
// MinimalRNN linear recurrence on gfx1250 (MI455X), chunked-scan formulation.
//   h_t = x_t @ W + h_{t-1} @ R     (B=32, T=4096, D=U=256)
// Serial depth cut 4096 -> ~192 via R^64 matrix-power boundary combine.
// f16 WMMA (v_wmma_f32_16x16x32_f16), weights resident in VGPRs as B-frags,
// ping-ponged LDS h/x staging -> ONE barrier per step, A-frag loads
// pipelined, and TWO independent WMMA accumulator chains (h@R, x@W) per step.
// ---------------------------------------------------------------------------

typedef __attribute__((ext_vector_type(16))) _Float16 v16h;
typedef __attribute__((ext_vector_type(8)))  _Float16 v8h;
typedef __attribute__((ext_vector_type(8)))  float    v8f;

#define B_    32
#define T_    4096
#define D_    256
#define U_    256
#define LCHUNK 64
#define NCHUNK 64           // T_/LCHUNK
#define XSTR  272           // padded LDS row stride in halves (16B aligned, bank-skewed)

// --- build the 8 B-fragments (K=256 split into 8 K-tiles of 32) for this
// wave's 16-column N-slice, from a row-major fp32 [K=256][N=256] matrix.
// B order used here (contiguous k, matches load_afrag): lanes 0-15 hold N=0..15 / K=kbase..kbase+15,
// lanes 16-31 same N with K=+16; VGPR v holds K = kbase+2v, kbase+2v+1.
__device__ __forceinline__ void load_bfrag(const float* __restrict__ M,
                                           int ncol0, v16h frag[8]) {
  const int lane  = threadIdx.x & 31;
  const int n     = ncol0 + (lane & 15);
  const int kbase = (lane >> 4) * 16;
#pragma unroll
  for (int kt = 0; kt < 8; ++kt) {
    v16h f;
#pragma unroll
    for (int i = 0; i < 16; ++i) {
      const int k = kt * 32 + kbase + i;
      f[i] = (_Float16)M[k * 256 + n];
    }
    frag[kt] = f;
  }
}

// --- A-fragment (16x32 f16 tile, K-tile kt) from an LDS f16 buffer,
// row-major [16][XSTR]. Lane m=l&15, K-halves
// {koff..koff+15} with koff = (l>>4)*16 -> two ds_load_b128.
__device__ __forceinline__ v16h load_afrag(const _Float16* buf, int kt) {
  const int lane = threadIdx.x & 31;
  const int m    = lane & 15;
  const int koff = (lane >> 4) * 16;
  const _Float16* base = buf + m * XSTR + kt * 32 + koff;
  v8h lo = *(const v8h*)(base);
  v8h hi = *(const v8h*)(base + 8);
  v16h a;
#pragma unroll
  for (int i = 0; i < 8; ++i) { a[i] = lo[i]; a[i + 8] = hi[i]; }
  return a;
}

// ---------------------------------------------------------------------------
// Generic chunked linear-scan kernel.
//   grid.x = #chunks (each runs L serial steps), grid.y = batch tile (16 rows)
//   HAS_W : h_t = x_t @ W + h_{t-1} @ R   (x is [.,.,D] via strides)
//   !HAS_W: h_t = x_t     + h_{t-1} @ R   (identity input projection; pass B)
// hinit0: fp32 [32][256] initial state for chunk 0 (nullptr -> zeros)
// hinitPrev: fp32 [NCHUNK][32][256]; chunk c>0 seeds from slice c-1 (nullptr->0)
// out: optional per-step output, element strides ob (batch) / ot (time)
// hfinal: optional fp32 [NCHUNK][32][256] chunk-final state dump
// ---------------------------------------------------------------------------
template <bool HAS_W>
__global__ __launch_bounds__(512) void scan_kernel(
    const float* __restrict__ xin, long xb, long xt,
    const float* __restrict__ Wmat, const float* __restrict__ Rmat,
    const float* __restrict__ hinit0, const float* __restrict__ hinitPrev,
    float* __restrict__ out, long ob, long ot,
    float* __restrict__ hfinal, int L) {
  // Ping-pong buffers: one barrier per serial step.
  __shared__ _Float16 xbuf[2][16 * XSTR];
  __shared__ _Float16 hbuf[2][16 * XSTR];

  const int tid    = threadIdx.x;
  const int wave   = tid >> 5;
  const int lane   = tid & 31;
  const int hi16   = lane >> 4;
  const int nloc   = lane & 15;
  const int ncol   = wave * 16 + nloc;
  const int chunk  = blockIdx.x;
  const int btbase = blockIdx.y * 16;

  // Resident weight B-fragments (loaded once, live in VGPRs all kernel).
  v16h Rf[8];
  load_bfrag(Rmat, wave * 16, Rf);
  v16h Wf[8];
  if (HAS_W) load_bfrag(Wmat, wave * 16, Wf);

  // Seed h state into hbuf[0] (f16).
  {
    const int row = tid >> 5;
    const int c0  = (tid & 31) * 8;
    const float* hp = nullptr;
    if (chunk == 0)        hp = hinit0;
    else if (hinitPrev)    hp = hinitPrev + (long)(chunk - 1) * 32 * 256;
    v8h hv;
    if (hp) {
      const float* p = hp + (long)(btbase + row) * 256 + c0;
#pragma unroll
      for (int i = 0; i < 8; ++i) hv[i] = (_Float16)p[i];
    } else {
#pragma unroll
      for (int i = 0; i < 8; ++i) hv[i] = (_Float16)0.0f;
    }
    *(v8h*)&hbuf[0][row * XSTR + c0] = hv;
  }

  const int t0 = chunk * L;
  v8f c = {};
  for (int s = 0; s < L; ++s) {
    const int t = t0 + s;
    const int p = s & 1;
    _Float16* xst = xbuf[p];

    // Stage x_t tile [16 rows][256] fp32 -> f16 into this step's x buffer.
    {
      const int row = tid >> 5;
      const int c0  = (tid & 31) * 8;
      const float* xp = xin + (long)(btbase + row) * xb + (long)t * xt + c0;
      const float4* xp4 = (const float4*)xp;
      float4 u0 = xp4[0];
      float4 u1 = xp4[1];
      v8h xv;
      xv[0] = (_Float16)u0.x; xv[1] = (_Float16)u0.y;
      xv[2] = (_Float16)u0.z; xv[3] = (_Float16)u0.w;
      xv[4] = (_Float16)u1.x; xv[5] = (_Float16)u1.y;
      xv[6] = (_Float16)u1.z; xv[7] = (_Float16)u1.w;
      *(v8h*)&xst[row * XSTR + c0] = xv;
      if (s + 1 < L) __builtin_prefetch(xp + xt, 0, 3);  // near-scope prefetch
    }
    __syncthreads();  // xbuf[p] staged; hbuf[p] (written at step s-1) visible

    const _Float16* hr = hbuf[p];
    _Float16*       hw = hbuf[p ^ 1];

    if (HAS_W) {
      // Two independent WMMA accumulator chains (h@R and x@W), interleaved,
      // with A-frag LDS loads pipelined one K-tile ahead of their WMMA.
      v8f c0 = {}, c1 = {};
      v16h ah = load_afrag(hr, 0);
      v16h ax = load_afrag(xst, 0);
#pragma unroll
      for (int kt = 0; kt < 8; ++kt) {
        v16h ah_n = ah, ax_n = ax;
        if (kt < 7) {
          ah_n = load_afrag(hr, kt + 1);
          ax_n = load_afrag(xst, kt + 1);
        }
        c0 = __builtin_amdgcn_wmma_f32_16x16x32_f16(false, ah, false, Rf[kt],
                                                    (short)0, c0, false, false);
        c1 = __builtin_amdgcn_wmma_f32_16x16x32_f16(false, ax, false, Wf[kt],
                                                    (short)0, c1, false, false);
        ah = ah_n;
        ax = ax_n;
      }
      c = c0 + c1;
    } else {
      v8f c0 = {};
      v16h ah = load_afrag(hr, 0);
#pragma unroll
      for (int kt = 0; kt < 8; ++kt) {
        v16h ah_n = ah;
        if (kt < 7) ah_n = load_afrag(hr, kt + 1);
        c0 = __builtin_amdgcn_wmma_f32_16x16x32_f16(false, ah, false, Rf[kt],
                                                    (short)0, c0, false, false);
        ah = ah_n;
      }
      // identity input projection: add staged x directly into C layout
#pragma unroll
      for (int r = 0; r < 8; ++r)
        c0[r] += (float)xst[(r + 8 * hi16) * XSTR + ncol];
      c = c0;
    }

    if (out) {
#pragma unroll
      for (int r = 0; r < 8; ++r)
        out[(long)(btbase + r + 8 * hi16) * ob + (long)t * ot + ncol] = c[r];
    }
    // Publish h_t into the opposite buffer; next step's barrier orders it.
#pragma unroll
    for (int r = 0; r < 8; ++r)
      hw[(r + 8 * hi16) * XSTR + ncol] = (_Float16)c[r];
  }

  if (hfinal) {
#pragma unroll
    for (int r = 0; r < 8; ++r)
      hfinal[((long)chunk * 32 + btbase + r + 8 * hi16) * 256 + ncol] = c[r];
  }
}

// ---------------------------------------------------------------------------
// Out = A @ A for 256x256 fp32 (repeated squaring for R^64). Tiny, fp32 VALU.
// ---------------------------------------------------------------------------
__global__ __launch_bounds__(256) void matsq256(const float* __restrict__ A,
                                                float* __restrict__ Out) {
  const int n   = blockIdx.x * 256 + threadIdx.x;   // 0..65535
  const int row = n >> 8;
  const int col = n & 255;
  float acc = 0.0f;
#pragma unroll 8
  for (int k = 0; k < 256; ++k) acc = fmaf(A[row * 256 + k], A[k * 256 + col], acc);
  Out[n] = acc;
}

extern "C" void kernel_launch(void* const* d_in, const int* in_sizes, int n_in,
                              void* d_out, int out_size, void* d_ws, size_t ws_size,
                              hipStream_t stream) {
  const float* x  = (const float*)d_in[0];   // [B,T,D]
  const float* h0 = (const float*)d_in[1];   // [B,U]
  const float* W  = (const float*)d_in[2];   // [D,U]
  const float* R  = (const float*)d_in[3];   // [U,U]
  float* out = (float*)d_out;                // [B,T,U]

  float* ws   = (float*)d_ws;
  float* powA = ws;                          // 65536
  float* powB = powA + 65536;                // 65536
  float* pA   = powB + 65536;                // [64][32][256] chunk partials
  float* hb   = pA + (long)NCHUNK * 32 * 256; // [64][32][256] boundary states

  // R^64 by repeated squaring (6 tiny GEMMs, serialized on stream).
  matsq256<<<256, 256, 0, stream>>>(R,    powA);  // R^2
  matsq256<<<256, 256, 0, stream>>>(powA, powB);  // R^4
  matsq256<<<256, 256, 0, stream>>>(powB, powA);  // R^8
  matsq256<<<256, 256, 0, stream>>>(powA, powB);  // R^16
  matsq256<<<256, 256, 0, stream>>>(powB, powA);  // R^32
  matsq256<<<256, 256, 0, stream>>>(powA, powB);  // R^64  -> powB
  const float* RL = powB;

  // Pass A: per-chunk partial scans from zero (chunk 0 seeded with h0),
  // emit chunk-final partials only. 64 x 2 = 128 workgroups in parallel.
  scan_kernel<true><<<dim3(NCHUNK, 2), 512, 0, stream>>>(
      x, (long)T_ * D_, (long)D_, W, R,
      h0, nullptr, nullptr, 0, 0, pA, LCHUNK);

  // Pass B: serial boundary combine, same recurrence with R := R^64 and
  // identity input projection over the 64 partials. 2 workgroups, depth 64.
  scan_kernel<false><<<dim3(1, 2), 512, 0, stream>>>(
      pA, (long)256, (long)32 * 256, nullptr, RL,
      nullptr, nullptr, hb, (long)256, (long)32 * 256, nullptr, NCHUNK);

  // Pass C: re-scan every chunk seeded with the correct boundary state,
  // streaming all h_t to d_out. 128 workgroups in parallel.
  scan_kernel<true><<<dim3(NCHUNK, 2), 512, 0, stream>>>(
      x, (long)T_ * D_, (long)D_, W, R,
      h0, hb, out, (long)T_ * U_, (long)U_, nullptr, LCHUNK);
}
